// MPNNModel_45320494907958
// MI455X (gfx1250) — matrix operand order repaired and checked by reading, not compiled, not run on hardware
//
#include <hip/hip_runtime.h>
#include <hip/hip_bf16.h>

#define N_NODES   20000
#define N_EDGES   200000
#define F_NODE    11
#define HIDDEN    32
#define N_GRAPHS  512
#define N_TARGETS 12

typedef __attribute__((ext_vector_type(16))) __bf16 v16bf;
typedef __attribute__((ext_vector_type(8)))  float  v8f;

union AF { v16bf v; unsigned u[8]; };

// bf16 truncation pack: lo -> [15:0], hi -> [31:16]
__device__ __forceinline__ unsigned pk_bf16(float lo, float hi) {
    unsigned a = __float_as_uint(lo), b = __float_as_uint(hi);
    return (b & 0xffff0000u) | (a >> 16);
}

// packed bf16 multiply (VOP3P V_PK_MUL_BF16)
__device__ __forceinline__ unsigned pkmul_bf16(unsigned a, unsigned b) {
    unsigned d;
    asm("v_pk_mul_bf16 %0, %1, %2" : "=v"(d) : "v"(a), "v"(b));
    return d;
}

// monotonic float<->uint encoding for atomicMax-based segment max
__device__ __forceinline__ unsigned fenc(float f) {
    unsigned s = __float_as_uint(f);
    return (s & 0x80000000u) ? ~s : (s | 0x80000000u);
}
__device__ __forceinline__ float fdec(unsigned u) {
    return __uint_as_float((u & 0x80000000u) ? (u ^ 0x80000000u) : ~u);
}
__device__ __forceinline__ float sigm(float x) { return 1.0f / (1.0f + expf(-x)); }

// ---------------------------------------------------------------------------
// Prep: permute w2 (in_ch*32 x 32) + b2 into WMMA B-fragment order (bf16 pairs).
// Fragment index: idx = kk*512 + nh*256 + j*32 + lane
//   lane L: N = L&15, Khalf = L>>4 ; dword j packs K = (j<4 ? 0 : 16) + 8*Khalf + 2(j&3), +1
//   Main steps kk < in_ch : B[k,o] = w2[(kk*32+o)*32 + k]
//   Bias step  kk == in_ch: B[k,o] = (k<in_ch) ? b2[k*32+o] : 0
// ---------------------------------------------------------------------------
__global__ void prep_bfrag_kernel(const float* __restrict__ w2,
                                  const float* __restrict__ b2,
                                  unsigned* __restrict__ out, int in_ch) {
    int idx = blockIdx.x * blockDim.x + threadIdx.x;
    int total = (in_ch + 1) * 512;
    if (idx >= total) return;
    int lane = idx & 31;
    int j    = (idx >> 5) & 7;
    int nh   = (idx >> 8) & 1;
    int kk   = idx >> 9;
    int o  = nh * 16 + (lane & 15);
    int k0 = ((j < 4) ? 0 : 16) + 8 * (lane >> 4) + 2 * (j & 3);
    float f0, f1;
    if (kk < in_ch) {
        f0 = w2[(kk * 32 + o) * 32 + k0];
        f1 = w2[(kk * 32 + o) * 32 + k0 + 1];
    } else {
        f0 = (k0     < in_ch) ? b2[(k0    ) * 32 + o] : 0.f;
        f1 = (k0 + 1 < in_ch) ? b2[(k0 + 1) * 32 + o] : 0.f;
    }
    out[idx] = pk_bf16(f0, f1);
}

__global__ void count_edges_kernel(const int* __restrict__ dst, float* __restrict__ cnt) {
    int e = blockIdx.x * blockDim.x + threadIdx.x;
    if (e < N_EDGES) atomicAdd(&cnt[dst[e]], 1.0f);
}

// ---------------------------------------------------------------------------
// Fused NNConv message kernel. One wave = 32 edges (two 16-row M-tiles) so
// every B fragment loaded from LDS feeds 2 WMMAs.
//   msg = (x_src (x) h) [32 x (IN_CH+1)*32] @ Wmat -> v_wmma_f32_16x16x32_bf16
//   A K-slice kk: A[e,k] = xs[e][kk] * h[e][k]    (one v_pk_mul_bf16 / dword)
//   Bias slice:   A[e,k] = xs[e][k]
// B table is brought in with GLOBAL_LOAD_ASYNC_TO_LDS_B128 (ASYNCcnt),
// overlapped with the staging phase; msg scattered via global_atomic_add_f32.
// ---------------------------------------------------------------------------
template <int IN_CH, int XSTRIDE>
__global__ __launch_bounds__(256)
void edge_msg_kernel(const float* __restrict__ x,
                     const int* __restrict__ src, const int* __restrict__ dst,
                     const float* __restrict__ ea,
                     const float* __restrict__ w1, const float* __restrict__ b1,
                     const unsigned* __restrict__ bfrag,
                     float* __restrict__ agg) {
    constexpr int NSTEPS  = IN_CH;          // main K-steps (i index)
    constexpr int NFRAGDW = (NSTEPS + 1) * 512;
    extern __shared__ unsigned smem[];
    unsigned* ldsB = smem;
    const int wid  = threadIdx.x >> 5;
    const int lane = threadIdx.x & 31;

    // per-wave region: h 32x32, xs 32x32, src/dst 32+32, ea 32x4  = 2240 dw
    float* wbase  = (float*)(smem + NFRAGDW) + wid * 2304;
    float* ldsH   = wbase;             // 32x32 f32
    float* ldsX   = wbase + 1024;      // 32x32 f32 (zero past IN_CH)
    int*   ldsSrc = (int*)(wbase + 2048);
    int*   ldsDst = ldsSrc + 32;
    float* ldsEA  = (float*)(ldsDst + 32);   // 32x4 f32

    // ---- Phase 1: kick off async B-table copy (memory -> LDS, no VGPR data)
    {
        unsigned ldsBase = (unsigned)(size_t)(void*)ldsB;
        for (int i = threadIdx.x * 4; i < NFRAGDW; i += 256 * 4) {
            unsigned           laddr = ldsBase + (unsigned)i * 4u;
            unsigned long long gaddr = (unsigned long long)(size_t)(bfrag + i);
            asm volatile("global_load_async_to_lds_b128 %0, %1, off"
                         :: "v"(laddr), "v"(gaddr) : "memory");
        }
    }

    const int  grp    = blockIdx.x * 8 + wid;   // 32-edge group per wave
    const int  e0     = grp * 32;
    const bool active = e0 < N_EDGES;

    // ---- Phase 2: staging work overlapped with the async DMA
    if (active) {
        int eg = e0 + lane;
        bool ok = eg < N_EDGES;
        ldsSrc[lane] = ok ? src[eg] : 0;
        ldsDst[lane] = ok ? dst[eg] : -1;
        float4 v = ok ? ((const float4*)ea)[eg] : make_float4(0.f, 0.f, 0.f, 0.f);
        ((float4*)ldsEA)[lane] = v;
    }
    __syncthreads();

    if (active) {
        // per-lane column of edge MLP: h[e][lane] = relu(b1 + ea.w1row)
        float w0 = w1[lane * 4 + 0], wA = w1[lane * 4 + 1];
        float wB = w1[lane * 4 + 2], wC = w1[lane * 4 + 3];
        float bb = b1[lane];
        for (int e2 = 0; e2 < 32; ++e2) {
            float hv = bb + ldsEA[e2 * 4 + 0] * w0 + ldsEA[e2 * 4 + 1] * wA
                          + ldsEA[e2 * 4 + 2] * wB + ldsEA[e2 * 4 + 3] * wC;
            ldsH[e2 * 32 + lane] = fmaxf(hv, 0.f);
            int s_ = ldsSrc[e2];
            ldsX[e2 * 32 + lane] = (lane < IN_CH) ? x[(size_t)s_ * XSTRIDE + lane] : 0.f;
        }
    }

    // ---- join DMA: each wave drains its own ASYNCcnt, then block barrier
    asm volatile("s_wait_asynccnt 0x0" ::: "memory");
    __syncthreads();

    if (!active) return;

    const int er   = lane & 15;   // M row within tile
    const int half = lane >> 4;   // K-half per A layout

    // Pre-packed bf16 H fragments for both M-tiles.
    // A-layout K map: element m -> k = (m<8 ? 0 : 16) + 8*half + (m&7)
    unsigned hpk[2][8];
#pragma unroll
    for (int t = 0; t < 2; ++t) {
        const float* hr = &ldsH[(t * 16 + er) * 32];
#pragma unroll
        for (int j = 0; j < 8; ++j) {
            int m0 = 2 * j, m1 = 2 * j + 1;
            int k0 = ((m0 < 8) ? 0 : 16) + 8 * half + (m0 & 7);
            int k1 = ((m1 < 8) ? 0 : 16) + 8 * half + (m1 & 7);
            hpk[t][j] = pk_bf16(hr[k0], hr[k1]);
        }
    }

    v8f c00 = {}; v8f c01 = {}; v8f c10 = {}; v8f c11 = {};
    for (int kk = 0; kk <= NSTEPS; ++kk) {
        AF a0, a1;
        if (kk < NSTEPS) {
            unsigned sb0 = __float_as_uint(ldsX[er * 32 + kk]);
            unsigned sb1 = __float_as_uint(ldsX[(16 + er) * 32 + kk]);
            unsigned sp0 = (sb0 & 0xffff0000u) | (sb0 >> 16);   // splat bf16
            unsigned sp1 = (sb1 & 0xffff0000u) | (sb1 >> 16);
#pragma unroll
            for (int j = 0; j < 8; ++j) {
                a0.u[j] = pkmul_bf16(hpk[0][j], sp0);
                a1.u[j] = pkmul_bf16(hpk[1][j], sp1);
            }
        } else {
#pragma unroll
            for (int j = 0; j < 8; ++j) {
                int m0 = 2 * j, m1 = 2 * j + 1;
                int k0 = ((m0 < 8) ? 0 : 16) + 8 * half + (m0 & 7);
                int k1 = ((m1 < 8) ? 0 : 16) + 8 * half + (m1 & 7);
                a0.u[j] = pk_bf16(ldsX[er * 32 + k0],        ldsX[er * 32 + k1]);
                a1.u[j] = pk_bf16(ldsX[(16 + er) * 32 + k0], ldsX[(16 + er) * 32 + k1]);
            }
        }
        const unsigned* bp = ldsB + kk * 512;
        AF b0, b1f;
#pragma unroll
        for (int j = 0; j < 8; ++j) {
            b0.u[j]  = bp[j * 32 + lane];
            b1f.u[j] = bp[256 + j * 32 + lane];
        }
        // Order so each consecutive WMMA shares one operand; set reuse hints.
        c00 = __builtin_amdgcn_wmma_f32_16x16x32_bf16(false, a0.v, false, b0.v,
                                                      (short)0, c00, true,  false);
        c01 = __builtin_amdgcn_wmma_f32_16x16x32_bf16(false, a0.v, false, b1f.v,
                                                      (short)0, c01, false, true);
        c11 = __builtin_amdgcn_wmma_f32_16x16x32_bf16(false, a1.v, false, b1f.v,
                                                      (short)0, c11, true,  false);
        c10 = __builtin_amdgcn_wmma_f32_16x16x32_bf16(false, a1.v, false, b0.v,
                                                      (short)0, c10, false, false);
    }

    // D layout: VGPR r -> M = r + 8*half, N = lane&15
#pragma unroll
    for (int r = 0; r < 8; ++r) {
        int M = r + 8 * half;
        int d0 = ldsDst[M];
        if (d0 >= 0) {
            atomicAdd(&agg[d0 * 32 + (lane & 15)],      c00[r]);
            atomicAdd(&agg[d0 * 32 + 16 + (lane & 15)], c01[r]);
        }
        int d1 = ldsDst[16 + M];
        if (d1 >= 0) {
            atomicAdd(&agg[d1 * 32 + (lane & 15)],      c10[r]);
            atomicAdd(&agg[d1 * 32 + 16 + (lane & 15)], c11[r]);
        }
    }
}

// x_next = relu(agg/max(cnt,1) + x@root + bias)
__global__ void node_update_kernel(const float* __restrict__ agg, const float* __restrict__ cnt,
                                   const float* __restrict__ x, int in_ch, int xstride,
                                   const float* __restrict__ root, const float* __restrict__ bias,
                                   float* __restrict__ xn) {
    int idx = blockIdx.x * blockDim.x + threadIdx.x;
    if (idx >= N_NODES * 32) return;
    int v = idx >> 5, o = idx & 31;
    float a = agg[idx] / fmaxf(cnt[v], 1.0f);
    float r = bias[o];
    for (int i = 0; i < in_ch; ++i) r += x[(size_t)v * xstride + i] * root[i * 32 + o];
    xn[idx] = fmaxf(a + r, 0.0f);
}

// ------------------------------ Set2Set ------------------------------------
__global__ void s2s_gates_kernel(const float* __restrict__ qstar, const float* __restrict__ h,
                                 const float* __restrict__ wih, const float* __restrict__ whh,
                                 const float* __restrict__ bih, const float* __restrict__ bhh,
                                 float* __restrict__ gates) {
    int idx = blockIdx.x * blockDim.x + threadIdx.x;
    if (idx >= N_GRAPHS * 128) return;
    int b = idx >> 7, u = idx & 127;
    float g = bih[u] + bhh[u];
    for (int j = 0; j < 64; ++j) g += qstar[b * 64 + j] * wih[u * 64 + j];
    for (int j = 0; j < 32; ++j) g += h[b * 32 + j] * whh[u * 32 + j];
    gates[idx] = g;
}

__global__ void s2s_lstm_kernel(const float* __restrict__ gates,
                                float* __restrict__ h, float* __restrict__ c) {
    int idx = blockIdx.x * blockDim.x + threadIdx.x;
    if (idx >= N_GRAPHS * 32) return;
    int b = idx >> 5, o = idx & 31;
    float gi = gates[b * 128 + o];
    float gf = gates[b * 128 + 32 + o];
    float gg = gates[b * 128 + 64 + o];
    float go = gates[b * 128 + 96 + o];
    float cc = sigm(gf) * c[idx] + sigm(gi) * tanhf(gg);
    c[idx] = cc;
    h[idx] = sigm(go) * tanhf(cc);
}

__global__ void fill_u32_kernel(unsigned* __restrict__ p, unsigned v, int n) {
    int i = blockIdx.x * blockDim.x + threadIdx.x;
    if (i < n) p[i] = v;
}

__global__ void s2s_edot_kernel(const float* __restrict__ x, const float* __restrict__ h,
                                const int* __restrict__ batch,
                                float* __restrict__ e, unsigned* __restrict__ menc) {
    int v = blockIdx.x * blockDim.x + threadIdx.x;
    if (v >= N_NODES) return;
    int b = batch[v];
    float acc = 0.f;
    for (int o = 0; o < 32; ++o) acc += x[v * 32 + o] * h[b * 32 + o];
    e[v] = acc;
    atomicMax(&menc[b], fenc(acc));
}

__global__ void s2s_attn1_kernel(const float* __restrict__ e, const unsigned* __restrict__ menc,
                                 const int* __restrict__ batch,
                                 float* __restrict__ a, float* __restrict__ s) {
    int v = blockIdx.x * blockDim.x + threadIdx.x;
    if (v >= N_NODES) return;
    int b = batch[v];
    float av = expf(e[v] - fdec(menc[b]));
    a[v] = av;
    atomicAdd(&s[b], av);
}

__global__ void s2s_attn2_kernel(const float* __restrict__ a, const float* __restrict__ s,
                                 const float* __restrict__ x, const int* __restrict__ batch,
                                 float* __restrict__ r) {
    int idx = blockIdx.x * blockDim.x + threadIdx.x;
    if (idx >= N_NODES * 32) return;
    int v = idx >> 5, o = idx & 31;
    int b = batch[v];
    float w = a[v] / (s[b] + 1e-16f);
    atomicAdd(&r[b * 32 + o], w * x[idx]);
}

__global__ void s2s_qstar_kernel(const float* __restrict__ h, const float* __restrict__ r,
                                 float* __restrict__ qstar) {
    int idx = blockIdx.x * blockDim.x + threadIdx.x;
    if (idx >= N_GRAPHS * 64) return;
    int b = idx >> 6, j = idx & 63;
    qstar[idx] = (j < 32) ? h[b * 32 + j] : r[b * 32 + (j - 32)];
}

__global__ void final_lin_kernel(const float* __restrict__ qstar,
                                 const float* __restrict__ lw, const float* __restrict__ lb,
                                 float* __restrict__ out) {
    int idx = blockIdx.x * blockDim.x + threadIdx.x;
    if (idx >= N_GRAPHS * N_TARGETS) return;
    int b = idx / N_TARGETS, t = idx % N_TARGETS;
    float acc = lb[t];
    for (int j = 0; j < 64; ++j) acc += qstar[b * 64 + j] * lw[t * 64 + j];
    out[idx] = acc;
}

// ---------------------------------------------------------------------------
extern "C" void kernel_launch(void* const* d_in, const int* in_sizes, int n_in,
                              void* d_out, int out_size, void* d_ws, size_t ws_size,
                              hipStream_t stream) {
    const float* x_in   = (const float*)d_in[0];
    const int*   src    = (const int*)d_in[1];
    const int*   dst    = src + N_EDGES;
    const float* ea     = (const float*)d_in[2];
    const int*   batch  = (const int*)d_in[3];
    const float* w1l[3]   = { (const float*)d_in[4],  (const float*)d_in[10], (const float*)d_in[16] };
    const float* b1l[3]   = { (const float*)d_in[5],  (const float*)d_in[11], (const float*)d_in[17] };
    const float* w2l[3]   = { (const float*)d_in[6],  (const float*)d_in[12], (const float*)d_in[18] };
    const float* b2l[3]   = { (const float*)d_in[7],  (const float*)d_in[13], (const float*)d_in[19] };
    const float* rootl[3] = { (const float*)d_in[8],  (const float*)d_in[14], (const float*)d_in[20] };
    const float* biasl[3] = { (const float*)d_in[9],  (const float*)d_in[15], (const float*)d_in[21] };
    const float* wih = (const float*)d_in[22];
    const float* whh = (const float*)d_in[23];
    const float* bih = (const float*)d_in[24];
    const float* bhh = (const float*)d_in[25];
    const float* lw  = (const float*)d_in[26];
    const float* lb  = (const float*)d_in[27];
    float* out = (float*)d_out;

    // workspace carve-up (256B aligned)
    char* w = (char*)d_ws;
    auto carve = [&](size_t bytes) { void* p = (void*)w; w += (bytes + 255) & ~(size_t)255; return p; };
    float*    xbufA = (float*)carve(N_NODES * 32 * sizeof(float));
    float*    xbufB = (float*)carve(N_NODES * 32 * sizeof(float));
    float*    agg   = (float*)carve(N_NODES * 32 * sizeof(float));
    float*    cnt   = (float*)carve(N_NODES * sizeof(float));
    unsigned* bf[3];
    for (int l = 0; l < 3; ++l) bf[l] = (unsigned*)carve(33 * 512 * sizeof(unsigned));
    float*    earr  = (float*)carve(N_NODES * sizeof(float));
    float*    aarr  = (float*)carve(N_NODES * sizeof(float));
    unsigned* menc  = (unsigned*)carve(N_GRAPHS * sizeof(unsigned));
    float*    sarr  = (float*)carve(N_GRAPHS * sizeof(float));
    float*    rarr  = (float*)carve(N_GRAPHS * 32 * sizeof(float));
    float*    hC    = (float*)carve(N_GRAPHS * 32 * sizeof(float));
    float*    cC    = (float*)carve(N_GRAPHS * 32 * sizeof(float));
    float*    qst   = (float*)carve(N_GRAPHS * 64 * sizeof(float));
    float*    gat   = (float*)carve(N_GRAPHS * 128 * sizeof(float));

    const int inch[3] = { F_NODE, HIDDEN, HIDDEN };

    // B-fragment prep (tiny)
    for (int l = 0; l < 3; ++l) {
        int total = (inch[l] + 1) * 512;
        prep_bfrag_kernel<<<(total + 255) / 256, 256, 0, stream>>>(w2l[l], b2l[l], bf[l], inch[l]);
    }

    // edge in-degree counts (same every layer)
    hipMemsetAsync(cnt, 0, N_NODES * sizeof(float), stream);
    count_edges_kernel<<<(N_EDGES + 255) / 256, 256, 0, stream>>>(dst, cnt);

    const int edgeBlocks = ((N_EDGES / 32) + 7) / 8;   // 8 waves x 32 edges per block
    const int nodeGrid   = (N_NODES * 32 + 255) / 256;

    // ---- layer 0 (in_ch=11) ----
    hipMemsetAsync(agg, 0, N_NODES * 32 * sizeof(float), stream);
    {
        size_t sh = (size_t)(F_NODE + 1) * 512 * 4 + 8 * 2304 * 4;
        edge_msg_kernel<F_NODE, F_NODE><<<edgeBlocks, 256, sh, stream>>>(
            x_in, src, dst, ea, w1l[0], b1l[0], bf[0], agg);
    }
    node_update_kernel<<<nodeGrid, 256, 0, stream>>>(agg, cnt, x_in, F_NODE, F_NODE,
                                                     rootl[0], biasl[0], xbufA);
    // ---- layer 1 ----
    hipMemsetAsync(agg, 0, N_NODES * 32 * sizeof(float), stream);
    {
        size_t sh = (size_t)(HIDDEN + 1) * 512 * 4 + 8 * 2304 * 4;
        edge_msg_kernel<HIDDEN, HIDDEN><<<edgeBlocks, 256, sh, stream>>>(
            xbufA, src, dst, ea, w1l[1], b1l[1], bf[1], agg);
    }
    node_update_kernel<<<nodeGrid, 256, 0, stream>>>(agg, cnt, xbufA, HIDDEN, HIDDEN,
                                                     rootl[1], biasl[1], xbufB);
    // ---- layer 2 ----
    hipMemsetAsync(agg, 0, N_NODES * 32 * sizeof(float), stream);
    {
        size_t sh = (size_t)(HIDDEN + 1) * 512 * 4 + 8 * 2304 * 4;
        edge_msg_kernel<HIDDEN, HIDDEN><<<edgeBlocks, 256, sh, stream>>>(
            xbufB, src, dst, ea, w1l[2], b1l[2], bf[2], agg);
    }
    node_update_kernel<<<nodeGrid, 256, 0, stream>>>(agg, cnt, xbufB, HIDDEN, HIDDEN,
                                                     rootl[2], biasl[2], xbufA);

    // ---- Set2Set ----
    hipMemsetAsync(hC, 0, N_GRAPHS * 32 * sizeof(float), stream);
    hipMemsetAsync(cC, 0, N_GRAPHS * 32 * sizeof(float), stream);
    hipMemsetAsync(qst, 0, N_GRAPHS * 64 * sizeof(float), stream);
    for (int t = 0; t < 3; ++t) {
        s2s_gates_kernel<<<(N_GRAPHS * 128 + 255) / 256, 256, 0, stream>>>(qst, hC, wih, whh, bih, bhh, gat);
        s2s_lstm_kernel<<<(N_GRAPHS * 32 + 255) / 256, 256, 0, stream>>>(gat, hC, cC);
        fill_u32_kernel<<<(N_GRAPHS + 255) / 256, 256, 0, stream>>>(menc, 0x007FFFFFu, N_GRAPHS); // enc(-inf)
        hipMemsetAsync(sarr, 0, N_GRAPHS * sizeof(float), stream);
        hipMemsetAsync(rarr, 0, N_GRAPHS * 32 * sizeof(float), stream);
        s2s_edot_kernel<<<(N_NODES + 255) / 256, 256, 0, stream>>>(xbufA, hC, batch, earr, menc);
        s2s_attn1_kernel<<<(N_NODES + 255) / 256, 256, 0, stream>>>(earr, menc, batch, aarr, sarr);
        s2s_attn2_kernel<<<(N_NODES * 32 + 255) / 256, 256, 0, stream>>>(aarr, sarr, xbufA, batch, rarr);
        s2s_qstar_kernel<<<(N_GRAPHS * 64 + 255) / 256, 256, 0, stream>>>(hC, rarr, qst);
    }
    final_lin_kernel<<<(N_GRAPHS * N_TARGETS + 255) / 256, 256, 0, stream>>>(qst, lw, lb, out);
}
